// GraphSAGE_32487132627547
// MI455X (gfx1250) — matrix operand order repaired and checked by reading, not compiled, not run on hardware
//
#include <hip/hip_runtime.h>

// ---------------------------------------------------------------------------
// GraphSAGE 2-layer forward for MI455X (gfx1250, wave32, WMMA).
//
// Roofline: edge scatter dominates (memory bound). We exploit linearity of
// segment_sum to project h (64ch) -> P = h @ w2_l (32ch) BEFORE scattering,
// halving the dominant edge traffic. P and R = h @ w2_r are produced by one
// fused WMMA pass (v_wmma_f32_16x16x32_f16, K=64 as two k-steps, f32 accum).
// ---------------------------------------------------------------------------

typedef __attribute__((ext_vector_type(16))) _Float16 v16h;
typedef __attribute__((ext_vector_type(8)))  _Float16 v8h;
typedef __attribute__((ext_vector_type(8)))  float    v8f;

__global__ void gs_zero_f32(float* p, long n) {
  long i = (long)blockIdx.x * blockDim.x + threadIdx.x;
  if (i < n) p[i] = 0.0f;
}

// Degree + layer-1 neighbor aggregation (2 channels): one thread per edge.
__global__ void gs_edge_deg_aggx(const long long* __restrict__ ei,
                                 const float* __restrict__ x,
                                 float* __restrict__ deg,
                                 float* __restrict__ agg1, int E) {
  int e = blockIdx.x * blockDim.x + threadIdx.x;
  if (e >= E) return;
  int s = (int)ei[e];
  int d = (int)ei[(long)E + e];
  atomicAdd(&deg[d], 1.0f);
  atomicAdd(&agg1[(long)d * 2 + 0], x[(long)s * 2 + 0]);
  atomicAdd(&agg1[(long)d * 2 + 1], x[(long)s * 2 + 1]);
}

// Layer 1: h = relu(agg/deg @ w1_l + b1 + x @ w1_r); K=2 so plain VALU.
// Stores h directly as f16 (WMMA A-operand precision). Pads tail rows with 0.
__global__ void gs_layer1(const float* __restrict__ x,
                          const float* __restrict__ agg1,
                          const float* __restrict__ deg,
                          const float* __restrict__ w1l,
                          const float* __restrict__ b1,
                          const float* __restrict__ w1r,
                          _Float16* __restrict__ h16, int n, int npad) {
  int i = blockIdx.x * blockDim.x + threadIdx.x;  // npad*64 threads
  if (i >= npad * 64) return;
  int node = i >> 6, c = i & 63;
  float v = 0.0f;
  if (node < n) {
    float dg = fmaxf(deg[node], 1.0f);
    float a0 = agg1[(long)node * 2 + 0] / dg;
    float a1 = agg1[(long)node * 2 + 1] / dg;
    float x0 = x[(long)node * 2 + 0];
    float x1 = x[(long)node * 2 + 1];
    v = a0 * w1l[c] + a1 * w1l[64 + c] + b1[c] + x0 * w1r[c] + x1 * w1r[64 + c];
    v = fmaxf(v, 0.0f);
  }
  h16[(long)node * 64 + c] = (_Float16)v;
}

// Pack [w2_l | w2_r] into one f16 B matrix, shape [K=64][Ncols=64].
__global__ void gs_pack_w2(const float* __restrict__ w2l,
                           const float* __restrict__ w2r,
                           _Float16* __restrict__ B) {
  int i = blockIdx.x * blockDim.x + threadIdx.x;  // 64*64
  if (i >= 64 * 64) return;
  int k = i >> 6, c = i & 63;
  float v = (c < 32) ? w2l[k * 32 + c] : w2r[k * 32 + (c - 32)];
  B[k * 64 + c] = (_Float16)v;
}

// Fused layer-2 GEMM: P = h @ w2_l, R = h @ w2_r via v_wmma_f32_16x16x32_f16.
// One wave per group of 16-node tiles; 4 output 16x16 tiles x 2 k-steps each.
// Full tiles use a branch-free store path (no per-row exec juggling).
__global__ void gs_wmma_proj(const _Float16* __restrict__ h16,
                             const _Float16* __restrict__ B,
                             float* __restrict__ P, float* __restrict__ R,
                             int n, int ntiles) {
  const int lane = threadIdx.x & 31;
  const int g = lane >> 4;     // lane group (ISA 16x16 f16 layout)
  const int m = lane & 15;     // row for A, column for B/C/D
  const int wave = (blockIdx.x * blockDim.x + threadIdx.x) >> 5;
  const int TPW = 8;           // node tiles per wave (amortize B loads)

  // B fragments: b[j][kc] with element b[2i+t] = B[kc*32 + g*16 + 2i+t][j*16 + m]
  v16h bf[4][2];
#pragma unroll
  for (int j = 0; j < 4; ++j)
#pragma unroll
    for (int kc = 0; kc < 2; ++kc)
#pragma unroll
      for (int t = 0; t < 16; ++t)
        bf[j][kc][t] = B[(kc * 32 + g * 16 + t) * 64 + j * 16 + m];

  int t0 = wave * TPW;
  int t1 = t0 + TPW;
  if (t1 > ntiles) t1 = ntiles;
  for (int tile = t0; tile < t1; ++tile) {   // uniform per wave: EXEC all-1s
    int nb = tile * 16;
    const _Float16* row = h16 + (long)(nb + m) * 64;
    // A fragments: lane (g,m): a[0..15]=K[g*16..g*16+15] (matches B)
    v16h a[2];
#pragma unroll
    for (int kc = 0; kc < 2; ++kc) {
      v8h lo = *(const v8h*)(row + kc * 32 + g * 16);
      v8h hi = *(const v8h*)(row + kc * 32 + g * 16 + 8);
#pragma unroll
      for (int t = 0; t < 8; ++t) { a[kc][t] = lo[t]; a[kc][t + 8] = hi[t]; }
    }
    const bool full = (nb + 16 <= n);  // wave-uniform
#pragma unroll
    for (int j = 0; j < 4; ++j) {
      v8f acc = {0.f, 0.f, 0.f, 0.f, 0.f, 0.f, 0.f, 0.f};
#pragma unroll
      for (int kc = 0; kc < 2; ++kc)
        acc = __builtin_amdgcn_wmma_f32_16x16x32_f16(
            false, a[kc], false, bf[j][kc], (short)0, acc, false, false);
      float* dst = (j < 2) ? P : R;
      long base = (long)(nb + 8 * g) * 32 + (j & 1) * 16 + m;
      if (full) {
        // D layout: M = r + 8*g, N = m; 8 unguarded b32 stores (clause-able)
#pragma unroll
        for (int r = 0; r < 8; ++r) dst[base + (long)r * 32] = acc[r];
      } else {
#pragma unroll
        for (int r = 0; r < 8; ++r) {
          int node = nb + r + 8 * g;
          if (node < n) dst[base + (long)r * 32] = acc[r];
        }
      }
    }
  }
}

// Scatter projected features: one EDGE per wave, one CHANNEL per lane (32ch).
// Coalesced 128B gather + 32 global_atomic_add_f32 per edge.
__global__ void gs_scatter_p(const long long* __restrict__ ei,
                             const float* __restrict__ P,
                             float* __restrict__ Pscat, int E) {
  int gid = blockIdx.x * blockDim.x + threadIdx.x;
  int e = gid >> 5;
  int lane = gid & 31;
  if (e >= E) return;
  int s = (int)ei[e];
  int d = (int)ei[(long)E + e];
  float v = P[(long)s * 32 + lane];
  atomicAdd(&Pscat[(long)d * 32 + lane], v);
}

// out = Pscat/deg + b2 + R
__global__ void gs_finalize(const float* __restrict__ Pscat,
                            const float* __restrict__ R,
                            const float* __restrict__ deg,
                            const float* __restrict__ b2,
                            float* __restrict__ out, int n) {
  long i = (long)blockIdx.x * blockDim.x + threadIdx.x;
  if (i >= (long)n * 32) return;
  int node = (int)(i >> 5), c = (int)(i & 31);
  out[i] = Pscat[i] / fmaxf(deg[node], 1.0f) + b2[c] + R[i];
}

extern "C" void kernel_launch(void* const* d_in, const int* in_sizes, int n_in,
                              void* d_out, int out_size, void* d_ws, size_t ws_size,
                              hipStream_t stream) {
  const float*     x   = (const float*)d_in[0];
  const long long* ei  = (const long long*)d_in[1];   // int64 edge_index [2,E]
  const float*     w1l = (const float*)d_in[2];
  const float*     b1  = (const float*)d_in[3];
  const float*     w1r = (const float*)d_in[4];
  const float*     w2l = (const float*)d_in[5];
  const float*     b2  = (const float*)d_in[6];
  const float*     w2r = (const float*)d_in[7];

  const int n    = in_sizes[0] / 2;   // nodes (IN_CH = 2)
  const int E    = in_sizes[1] / 2;   // edges
  const int npad = (n + 15) & ~15;

  // Workspace carve-out (256B aligned regions).
  char* ws = (char*)d_ws;
  size_t off = 0;
  auto take = [&](size_t bytes) -> void* {
    size_t o = (off + 255) & ~(size_t)255;
    off = o + bytes;
    return (void*)(ws + o);
  };
  float*    deg   = (float*)take((size_t)n * 4);            // [n]
  float*    agg1  = (float*)take((size_t)n * 8);            // [n,2]
  float*    Pscat = (float*)take((size_t)n * 32 * 4);       // [n,32]
  _Float16* h16   = (_Float16*)take((size_t)npad * 64 * 2); // [npad,64]
  _Float16* Bf    = (_Float16*)take((size_t)64 * 64 * 2);   // [64,64]
  float*    P     = (float*)take((size_t)n * 32 * 4);       // [n,32]
  float*    R     = (float*)take((size_t)n * 32 * 4);       // [n,32]

  // Zero the accumulator span (deg .. Pscat end) every call (graph-replay safe).
  long zn = (long)(((char*)Pscat - (char*)deg) / 4) + (long)n * 32;
  gs_zero_f32<<<(int)((zn + 255) / 256), 256, 0, stream>>>(deg, zn);

  gs_edge_deg_aggx<<<(E + 255) / 256, 256, 0, stream>>>(ei, x, deg, agg1, E);

  gs_pack_w2<<<16, 256, 0, stream>>>(w2l, w2r, Bf);

  gs_layer1<<<(npad * 64 + 255) / 256, 256, 0, stream>>>(x, agg1, deg, w1l, b1,
                                                         w1r, h16, n, npad);

  const int ntiles = npad / 16;
  const int waves  = (ntiles + 7) / 8;
  gs_wmma_proj<<<(waves * 32 + 127) / 128, 128, 0, stream>>>(h16, Bf, P, R, n,
                                                             ntiles);

  long sthreads = (long)E * 32;
  gs_scatter_p<<<(int)((sthreads + 255) / 256), 256, 0, stream>>>(ei, P, Pscat, E);

  long fthreads = (long)n * 32;
  gs_finalize<<<(int)((fthreads + 255) / 256), 256, 0, stream>>>(Pscat, R, deg,
                                                                 b2, (float*)d_out,
                                                                 n);
}
